// ClusterAttn_15109694947351
// MI455X (gfx1250) — compile-verified, matrix operand order checked by reading, not run on hardware
//
#include <hip/hip_runtime.h>
#include <hip/hip_bf16.h>

typedef __attribute__((ext_vector_type(16))) _Float16 v16h;
typedef __attribute__((ext_vector_type(8)))  _Float16 v8h;
typedef __attribute__((ext_vector_type(8)))  float    v8f;
typedef _Float16 h16;

#define NBATCH 2
#define NT     4096      // tokens per batch
#define ROWS   8192      // NBATCH*NT
#define NGT    32768     // group tokens per batch (NT*8)

// ---------------------------------------------------------------------------
// WMMA fragment helpers (wave32, CDNA5 ISA 7.12.2 16-bit layouts).
// A-fragment: lane(m = lane&15, h = lane>>4) holds k in [8h,8h+7] and
// [16+8h, 16+8h+7] -> two contiguous 16B runs => two b128 loads.
// B-fragments come from pre-packed tiles: 512 halves, lane-major, 32B/lane.
// ---------------------------------------------------------------------------
__device__ __forceinline__ v16h concat8(v8h lo, v8h hi) {
  return __builtin_shufflevector(lo, hi, 0, 1, 2, 3, 4, 5, 6, 7,
                                 8, 9, 10, 11, 12, 13, 14, 15);
}
__device__ __forceinline__ v16h load_frag_a(const h16* A, int lda, int lane) {
  int m = lane & 15, h = lane >> 4;
  const h16* p = A + (size_t)m * lda + 8 * h;
  v8h lo = *(const v8h*)p;
  v8h hi = *(const v8h*)(p + 16);
  return concat8(lo, hi);
}
__device__ __forceinline__ v16h load_frag_bp(const h16* tile, int lane) {
  const h16* p = tile + lane * 16;
  v8h lo = *(const v8h*)p;
  v8h hi = *(const v8h*)(p + 8);
  return concat8(lo, hi);
}
__device__ __forceinline__ v8f zero8() {
  v8f z;
#pragma unroll
  for (int i = 0; i < 8; ++i) z[i] = 0.f;
  return z;
}
#define WMMA_F16(a, b, c) \
  __builtin_amdgcn_wmma_f32_16x16x32_f16(false, (a), false, (b), (short)0, (c), false, false)

// packed-tile index for element (k, n) of a K x N row-major B matrix
__device__ __forceinline__ size_t pack_off(int k, int n, int N) {
  int nnt = N >> 4;
  int kc = k >> 5, kk = k & 31;
  int h = (kk >> 3) & 1;
  int i = (kk & 15) - 8 * h + ((kk >= 16) ? 8 : 0);
  int lane = (n & 15) + 16 * h;
  int jn = n >> 4;
  return ((size_t)(kc * nnt + jn) * 32 + lane) * 16 + i;
}
__device__ __forceinline__ void pack_b(const float* __restrict__ w, h16* __restrict__ dst,
                                       int N, int p) {
  int tile = p >> 9, lane = (p >> 4) & 31, i = p & 15;
  int nnt = N >> 4;
  int kc = tile / nnt, jn = tile - kc * nnt;
  int h = lane >> 4;
  int v = i >> 1, j = i & 1;
  int kk = 2 * v + j + 8 * h + ((v >= 4) ? 8 : 0);
  int k = kc * 32 + kk;
  int n = jn * 16 + (lane & 15);
  dst[p] = (h16)w[(size_t)k * N + n];
}

// ---------------------------------------------------------------------------
// K0: fp32 -> fp16 conversion with fragment packing of B operands
// ---------------------------------------------------------------------------
__global__ void k_pack(const float* __restrict__ we, const float* __restrict__ wc,
                       const float* __restrict__ wq, h16* __restrict__ wep,
                       h16* __restrict__ wcp, h16* __restrict__ wqp) {
  int p = blockIdx.x * 256 + threadIdx.x;
  if (p < 64 * 128)   pack_b(we, wep, 128, p);
  if (p < 128 * 1024) pack_b(wc, wcp, 1024, p);
  if (p < 64 * 64)    pack_b(wq, wqp, 64, p);
}

// ---------------------------------------------------------------------------
// K1: conv 96ch -> 1ch (SAME, 3x3x3) + patchify -> fea f16 (A operand)
// ---------------------------------------------------------------------------
__global__ void k_dwconv(const float* __restrict__ x, const float* __restrict__ w,
                         const float* __restrict__ bias, h16* __restrict__ fea) {
  __shared__ float wl[96 * 27];
  for (int i = threadIdx.x; i < 96 * 27; i += 256) wl[i] = w[i];
  __syncthreads();
  int v   = blockIdx.x * 256 + threadIdx.x;      // 524288 voxels
  int b   = v >> 18;
  int rem = v & 262143;
  int d = rem >> 12, hh = (rem >> 6) & 63, ww = rem & 63;
  float acc = bias[0];
  for (int kd = 0; kd < 3; ++kd) {
    int zz = d + kd - 1; if (zz < 0 || zz > 63) continue;
    for (int kh = 0; kh < 3; ++kh) {
      int yy = hh + kh - 1; if (yy < 0 || yy > 63) continue;
      for (int kw = 0; kw < 3; ++kw) {
        int xx = ww + kw - 1; if (xx < 0 || xx > 63) continue;
        int tap = kd * 9 + kh * 3 + kw;
        const float4* xp =
            (const float4*)(x + (size_t)(((b * 64 + zz) * 64 + yy) * 64 + xx) * 96);
#pragma unroll
        for (int c4 = 0; c4 < 24; ++c4) {
          float4 xv = xp[c4];
          int cb = c4 * 4;
          acc += xv.x * wl[(cb + 0) * 27 + tap] + xv.y * wl[(cb + 1) * 27 + tap] +
                 xv.z * wl[(cb + 2) * 27 + tap] + xv.w * wl[(cb + 3) * 27 + tap];
        }
      }
    }
  }
  int zb = d >> 2, zp = d & 3, yb = hh >> 2, yp = hh & 3, xb = ww >> 2, xpp = ww & 3;
  int n = (zb * 16 + yb) * 16 + xb;
  int f = (zp * 4 + yp) * 4 + xpp;
  fea[((size_t)(b * NT + n)) * 64 + f] = (h16)acc;
}

// ---------------------------------------------------------------------------
// K2: fea (8192x64) @ W_exp (64x128) + b  -> fea2 (f32 + f16)
// ---------------------------------------------------------------------------
__global__ void k_gemm_exp(const h16* __restrict__ fea, const h16* __restrict__ wp,
                           const float* __restrict__ bias, float* __restrict__ fea2,
                           h16* __restrict__ fea2h) {
  int wave = threadIdx.x >> 5, lane = threadIdx.x & 31;
  int m0 = blockIdx.x * 128 + wave * 16;
  v8f acc[8];
#pragma unroll
  for (int j = 0; j < 8; ++j) acc[j] = zero8();
#pragma unroll
  for (int kc = 0; kc < 2; ++kc) {
    v16h a = load_frag_a(fea + (size_t)m0 * 64 + kc * 32, 64, lane);
#pragma unroll
    for (int j = 0; j < 8; ++j) {
      v16h bm = load_frag_bp(wp + (size_t)(kc * 8 + j) * 512, lane);
      acc[j] = WMMA_F16(a, bm, acc[j]);
    }
  }
  int hh = lane >> 4, nn = lane & 15;
#pragma unroll
  for (int j = 0; j < 8; ++j)
#pragma unroll
    for (int r = 0; r < 8; ++r) {
      int m = m0 + r + 8 * hh, col = j * 16 + nn;
      float val = acc[j][r] + bias[col];
      fea2[(size_t)m * 128 + col]  = val;
      fea2h[(size_t)m * 128 + col] = (h16)val;
    }
}

// ---------------------------------------------------------------------------
// K2b: ga = sigmoid(fea2 @ W_ga + b), laid out as (b, token*8+g)
// ---------------------------------------------------------------------------
__global__ void k_ga(const float* __restrict__ fea2, const float* __restrict__ w,
                     const float* __restrict__ bias, float* __restrict__ ga) {
  int idx = blockIdx.x * 256 + threadIdx.x;     // ROWS*8
  int r = idx >> 3, g = idx & 7;
  const float* fp = fea2 + (size_t)r * 128;
  float acc = bias[g];
  for (int k = 0; k < 128; ++k) acc += fp[k] * w[k * 8 + g];
  float s = 1.f / (1.f + __expf(-acc));
  int b = r >> 12, tok = r & 4095;
  ga[(size_t)b * NGT + tok * 8 + g] = s;
}

// ---------------------------------------------------------------------------
// K3: fea2 (8192x128) @ cluster_weights (128x1024) -> act_raw f32
// ---------------------------------------------------------------------------
__global__ void k_gemm_cluster(const h16* __restrict__ a16, const h16* __restrict__ wp,
                               float* __restrict__ out) {
  int wave = threadIdx.x >> 5, lane = threadIdx.x & 31;
  int m0 = blockIdx.x * 16;
  int n0 = wave * 128;
  v8f acc[8];
#pragma unroll
  for (int j = 0; j < 8; ++j) acc[j] = zero8();
#pragma unroll
  for (int kc = 0; kc < 4; ++kc) {
    v16h a = load_frag_a(a16 + (size_t)m0 * 128 + kc * 32, 128, lane);
#pragma unroll
    for (int j = 0; j < 8; ++j) {
      v16h bm = load_frag_bp(wp + (size_t)(kc * 64 + wave * 8 + j) * 512, lane);
      acc[j] = WMMA_F16(a, bm, acc[j]);
    }
  }
  int hh = lane >> 4, nn = lane & 15;
#pragma unroll
  for (int j = 0; j < 8; ++j)
#pragma unroll
    for (int r = 0; r < 8; ++r)
      out[(size_t)(m0 + r + 8 * hh) * 1024 + n0 + j * 16 + nn] = acc[j][r];
}

// ---------------------------------------------------------------------------
// K4: per-column mean/rstd over 8192 rows (BN axis 0)
// ---------------------------------------------------------------------------
__global__ void k_colstats(const float* __restrict__ act, float* __restrict__ mean,
                           float* __restrict__ rstd) {
  __shared__ float s1[256], s2[256];
  int c = blockIdx.x, t = threadIdx.x;
  float a = 0.f, b = 0.f;
  for (int r = t; r < ROWS; r += 256) {
    float v = act[(size_t)r * 1024 + c];
    a += v; b += v * v;
  }
  s1[t] = a; s2[t] = b; __syncthreads();
  for (int s = 128; s > 0; s >>= 1) {
    if (t < s) { s1[t] += s1[t + s]; s2[t] += s2[t + s]; }
    __syncthreads();
  }
  if (t == 0) {
    float m = s1[0] / (float)ROWS;
    float v = s2[0] / (float)ROWS - m * m;
    mean[c] = m; rstd[c] = rsqrtf(v + 1e-5f);
  }
}

// ---------------------------------------------------------------------------
// K5: BN + softmax(128) * gate, in place on act_raw
// ---------------------------------------------------------------------------
__global__ void k_bnsoftmax(float* __restrict__ act, const float* __restrict__ mean,
                            const float* __restrict__ rstd, const float* __restrict__ g,
                            const float* __restrict__ be, const float* __restrict__ ga) {
  __shared__ float red[128];
  int chunk = blockIdx.x;             // r*8 + grp
  int r = chunk >> 3, grp = chunk & 7;
  int t = threadIdx.x;
  int col = grp * 128 + t;
  size_t idx = (size_t)r * 1024 + col;
  float v = (act[idx] - mean[col]) * rstd[col] * g[col] + be[col];
  red[t] = v; __syncthreads();
  for (int s = 64; s > 0; s >>= 1) {
    if (t < s) red[t] = fmaxf(red[t], red[t + s]);
    __syncthreads();
  }
  float mx = red[0]; __syncthreads();
  float e = __expf(v - mx);
  red[t] = e; __syncthreads();
  for (int s = 64; s > 0; s >>= 1) {
    if (t < s) red[t] += red[t + s];
    __syncthreads();
  }
  float sum = red[0];
  int b = r >> 12, gt = (r & 4095) * 8 + grp;
  act[idx] = e / sum * ga[(size_t)b * NGT + gt];
}

// ---------------------------------------------------------------------------
// K5b: cent[b,c,f] = sum_n act[b,n,c] * fea2g[b,n,f]   (n=32768, f=16)
// ---------------------------------------------------------------------------
__global__ void k_cent(const float* __restrict__ act, const float* __restrict__ fea2,
                       float* __restrict__ cent) {
  __shared__ float red[256 * 16];
  int b = blockIdx.x >> 7, c = blockIdx.x & 127;
  int t = threadIdx.x;
  float acc[16];
#pragma unroll
  for (int f = 0; f < 16; ++f) acc[f] = 0.f;
  for (int n = t; n < NGT; n += 256) {
    int r = b * 4096 + (n >> 3);
    int g = n & 7;
    float a = act[(size_t)r * 1024 + g * 128 + c];
    const float* fp = fea2 + (size_t)r * 128 + g * 16;
#pragma unroll
    for (int f = 0; f < 16; ++f) acc[f] += a * fp[f];
  }
#pragma unroll
  for (int f = 0; f < 16; ++f) red[t * 16 + f] = acc[f];
  __syncthreads();
  for (int s = 128; s > 0; s >>= 1) {
    if (t < s)
      for (int f = 0; f < 16; ++f) red[t * 16 + f] += red[(t + s) * 16 + f];
    __syncthreads();
  }
  if (t < 16) cent[((size_t)b * 128 + c) * 16 + t] = red[t];
}

// ---------------------------------------------------------------------------
// K6: cent @ proj + BN(axis 0,2) + kv.  Writes k^T and v already in
// fragment-packed B-tile order (k^T: K=64,N=128 ; v: K=128,N=64).
// ---------------------------------------------------------------------------
__global__ void k_centproj_kv(const float* __restrict__ cent, const float* __restrict__ pw,
                              const float* __restrict__ pb, const float* __restrict__ pg,
                              const float* __restrict__ pbeta, const float* __restrict__ kvw,
                              const float* __restrict__ kvb, float* __restrict__ centp,
                              h16* __restrict__ kTp, h16* __restrict__ vp) {
  __shared__ float scale_s[128], shift_s[128];
  int t = threadIdx.x;
  for (int idx = t; idx < 2 * 128 * 64; idx += 256) {
    int f = idx & 63, bc = idx >> 6;
    const float* cp = cent + (size_t)bc * 16;
    float a = pb[f];
    for (int g2 = 0; g2 < 16; ++g2) a += cp[g2] * pw[g2 * 64 + f];
    centp[idx] = a;
  }
  __syncthreads();
  if (t < 128) {
    float m = 0.f, v2 = 0.f;
    for (int b = 0; b < 2; ++b)
      for (int f = 0; f < 64; ++f) {
        float xv = centp[((size_t)b * 128 + t) * 64 + f];
        m += xv; v2 += xv * xv;
      }
    m /= 128.f; v2 = v2 / 128.f - m * m;
    float sc = rsqrtf(v2 + 1e-5f) * pg[t];
    scale_s[t] = sc;
    shift_s[t] = pbeta[t] - m * sc;
  }
  __syncthreads();
  for (int idx = t; idx < 2 * 128 * 128; idx += 256) {
    int j = idx & 127, bc = idx >> 7;
    int b = bc >> 7, c = bc & 127;
    const float* cp = centp + (size_t)bc * 64;
    float sc = scale_s[c], sh = shift_s[c];
    float a = kvb[j];
    for (int f = 0; f < 64; ++f) a += (cp[f] * sc + sh) * kvw[f * 128 + j];
    if (j < 64) kTp[(size_t)b * 8192 + pack_off(j, c, 128)] = (h16)a;        // B of QK^T
    else        vp [(size_t)b * 8192 + pack_off(c, j - 64, 64)] = (h16)a;    // B of PV
  }
}

// ---------------------------------------------------------------------------
// K7: q = fea @ W_q + b  -> f16 (A operand of attention)
// ---------------------------------------------------------------------------
__global__ void k_gemm_q(const h16* __restrict__ fea, const h16* __restrict__ wp,
                         const float* __restrict__ bias, h16* __restrict__ q16) {
  int wave = threadIdx.x >> 5, lane = threadIdx.x & 31;
  int m0 = blockIdx.x * 128 + wave * 16;
  v8f acc[4];
#pragma unroll
  for (int j = 0; j < 4; ++j) acc[j] = zero8();
#pragma unroll
  for (int kc = 0; kc < 2; ++kc) {
    v16h a = load_frag_a(fea + (size_t)m0 * 64 + kc * 32, 64, lane);
#pragma unroll
    for (int j = 0; j < 4; ++j) {
      v16h bm = load_frag_bp(wp + (size_t)(kc * 4 + j) * 512, lane);
      acc[j] = WMMA_F16(a, bm, acc[j]);
    }
  }
  int hh = lane >> 4, nn = lane & 15;
#pragma unroll
  for (int j = 0; j < 4; ++j)
#pragma unroll
    for (int r = 0; r < 8; ++r) {
      int m = m0 + r + 8 * hh, col = j * 16 + nn;
      q16[(size_t)m * 64 + col] = (h16)(acc[j][r] + bias[col]);
    }
}

// ---------------------------------------------------------------------------
// K8: attention, 32 q-rows/block: S = QK^T/8, softmax, O = P V -> 3D scatter
// ---------------------------------------------------------------------------
__global__ void k_attn(const h16* __restrict__ q16, const h16* __restrict__ kTp,
                       const h16* __restrict__ vp, float* __restrict__ no) {
  __shared__ float S[32 * 128];
  __shared__ h16   Sh[32 * 128];
  int b = blockIdx.x >> 7, rblk = blockIdx.x & 127;
  int t = threadIdx.x;            // 64 threads = 2 waves
  int wave = t >> 5, lane = t & 31;
  int mloc = wave * 16;
  int row0 = rblk * 32;
  const h16* kb = kTp + (size_t)b * 8192;
  const h16* vb = vp + (size_t)b * 8192;
  {
    v8f acc[8];
#pragma unroll
    for (int j = 0; j < 8; ++j) acc[j] = zero8();
#pragma unroll
    for (int kc = 0; kc < 2; ++kc) {
      v16h a = load_frag_a(q16 + ((size_t)b * NT + row0 + mloc) * 64 + kc * 32, 64, lane);
#pragma unroll
      for (int j = 0; j < 8; ++j) {
        v16h bm = load_frag_bp(kb + (size_t)(kc * 8 + j) * 512, lane);
        acc[j] = WMMA_F16(a, bm, acc[j]);
      }
    }
    int hh = lane >> 4, nn = lane & 15;
#pragma unroll
    for (int j = 0; j < 8; ++j)
#pragma unroll
      for (int r = 0; r < 8; ++r)
        S[(mloc + r + 8 * hh) * 128 + j * 16 + nn] = acc[j][r] * 0.125f;
  }
  __syncthreads();
  if (t < 32) {
    float mx = -1e30f;
    for (int c = 0; c < 128; ++c) mx = fmaxf(mx, S[t * 128 + c]);
    float sum = 0.f;
    for (int c = 0; c < 128; ++c) {
      float e = __expf(S[t * 128 + c] - mx);
      S[t * 128 + c] = e; sum += e;
    }
    float inv = 1.f / sum;
    for (int c = 0; c < 128; ++c) Sh[t * 128 + c] = (h16)(S[t * 128 + c] * inv);
  }
  __syncthreads();
  {
    v8f acc[4];
#pragma unroll
    for (int j = 0; j < 4; ++j) acc[j] = zero8();
#pragma unroll
    for (int kc = 0; kc < 4; ++kc) {
      v16h a = load_frag_a(Sh + mloc * 128 + kc * 32, 128, lane);
#pragma unroll
      for (int j = 0; j < 4; ++j) {
        v16h bm = load_frag_bp(vb + (size_t)(kc * 4 + j) * 512, lane);
        acc[j] = WMMA_F16(a, bm, acc[j]);
      }
    }
    int hh = lane >> 4, nn = lane & 15;
#pragma unroll
    for (int j = 0; j < 4; ++j)
#pragma unroll
      for (int r = 0; r < 8; ++r) {
        int n = row0 + mloc + r + 8 * hh;
        int f = j * 16 + nn;
        int xb = n & 15, yb = (n >> 4) & 15, zb = n >> 8;
        int xp = f & 3, yp = (f >> 2) & 3, zp = f >> 4;
        int d = zb * 4 + zp, hh2 = yb * 4 + yp, w2 = xb * 4 + xp;
        no[(((size_t)b * 64 + d) * 64 + hh2) * 64 + w2] = acc[j][r];
      }
  }
}

// ---------------------------------------------------------------------------
// K9: up-conv 1ch -> 96ch (SAME 3x3x3) + residual, channels-last output
// ---------------------------------------------------------------------------
__global__ void k_upconv(const float* __restrict__ no, const float* __restrict__ w,
                         const float* __restrict__ bias, const float* __restrict__ x,
                         float* __restrict__ out) {
  __shared__ float wl[96 * 27];
  for (int i = threadIdx.x; i < 96 * 27; i += 256) wl[i] = w[i];
  __syncthreads();
  size_t gid = (size_t)blockIdx.x * 256 + threadIdx.x;
  int chunk = (int)(gid % 24);
  size_t voxel = gid / 24;
  int b = (int)(voxel >> 18);
  int rem = (int)(voxel & 262143);
  int d = rem >> 12, hh = (rem >> 6) & 63, ww = rem & 63;
  int c0 = chunk * 4;
  float4 acc = make_float4(bias[c0], bias[c0 + 1], bias[c0 + 2], bias[c0 + 3]);
  for (int kd = 0; kd < 3; ++kd) {
    int zz = d + kd - 1; if (zz < 0 || zz > 63) continue;
    for (int kh = 0; kh < 3; ++kh) {
      int yy = hh + kh - 1; if (yy < 0 || yy > 63) continue;
      for (int kw = 0; kw < 3; ++kw) {
        int xx = ww + kw - 1; if (xx < 0 || xx > 63) continue;
        float nv = no[(((size_t)b * 64 + zz) * 64 + yy) * 64 + xx];
        int tap = kd * 9 + kh * 3 + kw;
        acc.x += nv * wl[(c0 + 0) * 27 + tap];
        acc.y += nv * wl[(c0 + 1) * 27 + tap];
        acc.z += nv * wl[(c0 + 2) * 27 + tap];
        acc.w += nv * wl[(c0 + 3) * 27 + tap];
      }
    }
  }
  size_t base = voxel * 96 + c0;
  float4 xv = *(const float4*)(x + base);
  float4 o = make_float4(acc.x + xv.x, acc.y + xv.y, acc.z + xv.z, acc.w + xv.w);
  *(float4*)(out + base) = o;
}

// ---------------------------------------------------------------------------
extern "C" void kernel_launch(void* const* d_in, const int* in_sizes, int n_in,
                              void* d_out, int out_size, void* d_ws, size_t ws_size,
                              hipStream_t stream) {
  (void)in_sizes; (void)n_in; (void)out_size; (void)ws_size;
  const float* x        = (const float*)d_in[0];
  const float* dwc_w    = (const float*)d_in[1];
  const float* dwc_b    = (const float*)d_in[2];
  const float* upc_w    = (const float*)d_in[3];
  const float* upc_b    = (const float*)d_in[4];
  const float* fc_exp_w = (const float*)d_in[5];
  const float* fc_exp_b = (const float*)d_in[6];
  const float* fc_ga_w  = (const float*)d_in[7];
  const float* fc_ga_b  = (const float*)d_in[8];
  const float* clus_w   = (const float*)d_in[9];
  const float* abn_g    = (const float*)d_in[10];
  const float* abn_b    = (const float*)d_in[11];
  const float* proj_w   = (const float*)d_in[12];
  const float* proj_b   = (const float*)d_in[13];
  const float* pbn_g    = (const float*)d_in[14];
  const float* pbn_b    = (const float*)d_in[15];
  const float* q_w      = (const float*)d_in[16];
  const float* q_b      = (const float*)d_in[17];
  const float* kv_w     = (const float*)d_in[18];
  const float* kv_b     = (const float*)d_in[19];
  float* out = (float*)d_out;

  char* ws = (char*)d_ws;
  size_t off = 0;
  auto take = [&](size_t bytes) -> void* {
    void* p = ws + off;
    off += (bytes + 255) & ~(size_t)255;
    return p;
  };
  h16*   fea16  = (h16*)  take((size_t)ROWS * 64 * 2);
  float* fea2f  = (float*)take((size_t)ROWS * 128 * 4);
  h16*   fea2h  = (h16*)  take((size_t)ROWS * 128 * 2);
  float* ga     = (float*)take((size_t)2 * NGT * 4);
  float* act    = (float*)take((size_t)ROWS * 1024 * 4);
  float* mean   = (float*)take(1024 * 4);
  float* rstd   = (float*)take(1024 * 4);
  float* cent   = (float*)take((size_t)2 * 128 * 16 * 4);
  float* centp  = (float*)take((size_t)2 * 128 * 64 * 4);
  h16*   kTp    = (h16*)  take((size_t)2 * 8192 * 2);
  h16*   vp     = (h16*)  take((size_t)2 * 8192 * 2);
  h16*   q16    = (h16*)  take((size_t)ROWS * 64 * 2);
  float* no     = (float*)take((size_t)2 * 64 * 64 * 64 * 4);
  h16*   wep    = (h16*)  take((size_t)64 * 128 * 2);
  h16*   wcp    = (h16*)  take((size_t)128 * 1024 * 2);
  h16*   wqp    = (h16*)  take((size_t)64 * 64 * 2);

  k_pack<<<512, 256, 0, stream>>>(fc_exp_w, clus_w, q_w, wep, wcp, wqp);
  k_dwconv<<<2048, 256, 0, stream>>>(x, dwc_w, dwc_b, fea16);
  k_gemm_exp<<<64, 256, 0, stream>>>(fea16, wep, fc_exp_b, fea2f, fea2h);
  k_ga<<<256, 256, 0, stream>>>(fea2f, fc_ga_w, fc_ga_b, ga);
  k_gemm_cluster<<<512, 256, 0, stream>>>(fea2h, wcp, act);
  k_colstats<<<1024, 256, 0, stream>>>(act, mean, rstd);
  k_bnsoftmax<<<65536, 128, 0, stream>>>(act, mean, rstd, abn_g, abn_b, ga);
  k_cent<<<256, 256, 0, stream>>>(act, fea2f, cent);
  k_centproj_kv<<<1, 256, 0, stream>>>(cent, proj_w, proj_b, pbn_g, pbn_b, kv_w, kv_b,
                                       centp, kTp, vp);
  k_gemm_q<<<64, 256, 0, stream>>>(fea16, wqp, q_b, q16);
  k_attn<<<256, 64, 0, stream>>>(q16, kTp, vp, no);
  k_upconv<<<49152, 256, 0, stream>>>(no, upc_w, upc_b, x, out);
}
